// SwinTransformerBlock3D_3607772529272
// MI455X (gfx1250) — matrix operand order repaired and checked by reading, not compiled, not run on hardware
//
#include <hip/hip_runtime.h>
#include <hip/hip_bf16.h>

// ---------------------------------------------------------------------------
// Swin Transformer 3D block for MI455X (gfx1250), wave32 + WMMA f16 (f32 acc).
// B=2, D=16, H=W=112, C=96, window (2,7,7) -> 98 tokens, shift (1,3,3),
// heads=3, head_dim=32, hidden=384.
// ---------------------------------------------------------------------------

typedef __attribute__((ext_vector_type(16))) _Float16 v16h;
typedef __attribute__((ext_vector_type(8)))  _Float16 v8h;
typedef __attribute__((ext_vector_type(8)))  float    v8f;

#define BATCH     2
#define DD        16
#define HH        112
#define WW        112
#define CC        96
#define NTOK      98
#define HEADS     3
#define HDIM      32
#define NWIN      4096          // total windows (B_)
#define NWB       2048          // windows per batch (mask count)
#define MTOT      (NWIN * NTOK) // 401408 rows, divisible by 16
#define MTILES    (MTOT / 16)   // 25088
#define HIDDEN    384
#define QSCALE    0.1767766952966369f   // 32^-0.5

// ---- workspace layout (bytes). total ~463 MB ------------------------------
#define OFF_WQKV  ((size_t)0)                         //  288*96 f16
#define OFF_WPROJ ((size_t)55296)                     //   96*96 f16
#define OFF_WFC1  ((size_t)(55296+18432))             //  384*96 f16
#define OFF_WFC2  ((size_t)(55296+18432+73728))       //  96*384 f16
#define OFF_XW    ((size_t)221184)                    // MTOT*96 f16 (reused: attn_out)
#define OFF_Q     (OFF_XW + (size_t)77070336)         // 4096*3*98*32 f16 (reused: ln2 out)
#define OFF_K     (OFF_Q  + (size_t)77070336)
#define OFF_V     (OFF_K  + (size_t)77070336)
#define OFF_X2    (OFF_V  + (size_t)77070336)         // MTOT*96 f32

// ---------------------------------------------------------------------------
__device__ __forceinline__ v8f wmma16(v16h a, v16h b, v8f c) {
  return __builtin_amdgcn_wmma_f32_16x16x32_f16(false, a, false, b, (short)0, c,
                                                false, false);
}

// A-tile (16 x 32 f16) from row-major f16 [row][stride]. lane(g,m): e<8 -> K=g*8+e,
// e>=8 -> K=16+g*8+(e-8). Two contiguous 16B loads.
__device__ __forceinline__ v16h load_A_g(const _Float16* base, int stride, int k0,
                                         int lane) {
  int m = lane & 15, g = lane >> 4;
  const _Float16* r = base + (size_t)m * stride + k0 + g * 8;
  v8h lo = *(const v8h*)(r);
  v8h hi = *(const v8h*)(r + 16);
  v16h out;
#pragma unroll
  for (int e = 0; e < 8; ++e) { out[e] = lo[e]; out[e + 8] = hi[e]; }
  return out;
}

// B-tile (32 x 16 f16) where column n is contiguous in memory: weights stored
// row-major [out_ch][K]. lane(g,n): e<8 -> K = k0 + g*8 + e, e>=8 -> K = k0 + 16 + g*8 + (e-8).
__device__ __forceinline__ v16h load_B_w(const _Float16* w, int K, int col, int k0,
                                         int lane) {
  int n = lane & 15, g = lane >> 4;
  const _Float16* p = w + (size_t)(col + n) * K + k0 + g * 8;
  v8h lo = *(const v8h*)(p);
  v8h hi = *(const v8h*)(p + 16);
  v16h out;
#pragma unroll
  for (int e = 0; e < 8; ++e) { out[e] = lo[e]; out[e + 8] = hi[e]; }
  return out;
}

// window-token (t, channel c) -> flat index into (B,D,H,W,C) applying the
// shifted-window gather/scatter mapping o = (s + SS) mod size.
__device__ __forceinline__ int win_token_to_addr(int t, int c) {
  int win = t / NTOK, n = t % NTOK;
  int b = win >> 11, r = win & 2047;
  int zd = r >> 8, yh = (r >> 4) & 15, xwi = r & 15;
  int dz = n / 49, rem = n % 49, hy = rem / 7, wx = rem % 7;
  int od = zd * 2 + dz + 1;  if (od >= DD) od -= DD;
  int oh = yh * 7 + hy + 3;  if (oh >= HH) oh -= HH;
  int ow = xwi * 7 + wx + 3; if (ow >= WW) ow -= WW;
  return (((b * DD + od) * HH + oh) * WW + ow) * CC + c;
}

// ---------------------------------------------------------------------------
// 1) weight f32 -> f16 conversion
__global__ __launch_bounds__(256) void k_cvt(const float* qkv_w, const float* proj_w,
                                             const float* fc1_w, const float* fc2_w,
                                             _Float16* wq, _Float16* wp,
                                             _Float16* w1, _Float16* w2) {
  int i = blockIdx.x * 256 + threadIdx.x;
  if (i < 288 * 96) wq[i] = (_Float16)qkv_w[i];
  if (i <  96 * 96) wp[i] = (_Float16)proj_w[i];
  if (i < 384 * 96) { w1[i] = (_Float16)fc1_w[i]; w2[i] = (_Float16)fc2_w[i]; }
}

// 2) LN1 + cyclic shift + window partition, f32 -> f16. One wave per token.
__global__ __launch_bounds__(256) void k_ln1(const float* __restrict__ x,
                                             const float* __restrict__ g1,
                                             const float* __restrict__ b1,
                                             _Float16* __restrict__ xw) {
  int wave = threadIdx.x >> 5, lane = threadIdx.x & 31;
  int t = blockIdx.x * 8 + wave;
  int base = win_token_to_addr(t, 0);
  const float* src = x + base;
  float v0 = src[lane], v1 = src[lane + 32], v2 = src[lane + 64];
  float s = v0 + v1 + v2, ss = v0 * v0 + v1 * v1 + v2 * v2;
#pragma unroll
  for (int off = 16; off; off >>= 1) {
    s  += __shfl_xor(s,  off, 32);
    ss += __shfl_xor(ss, off, 32);
  }
  float mu = s * (1.0f / 96.0f);
  float rinv = rsqrtf(ss * (1.0f / 96.0f) - mu * mu + 1e-5f);
  _Float16* dst = xw + (size_t)t * CC;
  dst[lane]      = (_Float16)((v0 - mu) * rinv * g1[lane]      + b1[lane]);
  dst[lane + 32] = (_Float16)((v1 - mu) * rinv * g1[lane + 32] + b1[lane + 32]);
  dst[lane + 64] = (_Float16)((v2 - mu) * rinv * g1[lane + 64] + b1[lane + 64]);
}

// 3) QKV GEMM: (MTOT x 96) x (96 x 288). Wave = 16-row tile, all 18 col tiles.
__global__ __launch_bounds__(256) void k_qkv(const _Float16* __restrict__ xw,
                                             const _Float16* __restrict__ w16,
                                             const float* __restrict__ qkv_b,
                                             _Float16* __restrict__ q,
                                             _Float16* __restrict__ k,
                                             _Float16* __restrict__ v) {
  int wave = threadIdx.x >> 5, lane = threadIdx.x & 31;
  int mt = blockIdx.x * 8 + wave;
  int row0 = mt * 16;
  v8f acc[18] = {};
#pragma unroll
  for (int ks = 0; ks < 3; ++ks) {
    v16h a = load_A_g(xw + (size_t)row0 * CC, CC, ks * 32, lane);
#pragma unroll
    for (int nt = 0; nt < 18; ++nt)
      acc[nt] = wmma16(a, load_B_w(w16, 96, nt * 16, ks * 32, lane), acc[nt]);
  }
  int g = lane >> 4, nc = lane & 15;
#pragma unroll
  for (int nt = 0; nt < 18; ++nt) {
    int col = nt * 16 + nc;
    int which = col / 96;           // 0=q 1=k 2=v
    int head = (col % 96) >> 5;
    int d = col & 31;
    float bias = qkv_b[col];
    float scl = (which == 0) ? QSCALE : 1.0f;
    _Float16* dst = (which == 0) ? q : ((which == 1) ? k : v);
#pragma unroll
    for (int vv = 0; vv < 8; ++vv) {
      int t = row0 + vv + g * 8;
      int win = t / NTOK, n = t % NTOK;
      dst[((size_t)(win * HEADS + head) * NTOK + n) * HDIM + d] =
          (_Float16)((acc[nt][vv] + bias) * scl);
    }
  }
}

// 4) attention per (window, head): S = qk^T + rpb + mask, softmax, P@v.
__global__ __launch_bounds__(256) void k_attn(const _Float16* __restrict__ qg,
                                              const _Float16* __restrict__ kg,
                                              const _Float16* __restrict__ vg,
                                              const float* __restrict__ rpb_table,
                                              const float* __restrict__ mask,
                                              _Float16* __restrict__ out) {
  __shared__ _Float16 S[112][128];    // scores -> probs (f16)
  __shared__ _Float16 qs[112][32];
  __shared__ _Float16 ksm[112][32];
  __shared__ _Float16 vs[128][32];

  int wh = blockIdx.x;
  int win = wh / HEADS, head = wh % HEADS;
  size_t hb = (size_t)(win * HEADS + head) * NTOK * HDIM;
  const _Float16* qb = qg + hb;
  const _Float16* kb = kg + hb;
  const _Float16* vb = vg + hb;
  int tid = threadIdx.x;

  for (int i = tid; i < 112 * 32; i += 256) {
    int r = i >> 5, c = i & 31;
    _Float16 zq = (r < NTOK) ? qb[r * 32 + c] : (_Float16)0.0f;
    _Float16 zk = (r < NTOK) ? kb[r * 32 + c] : (_Float16)0.0f;
    qs[r][c] = zq; ksm[r][c] = zk;
  }
  for (int i = tid; i < 128 * 32; i += 256) {
    int r = i >> 5, c = i & 31;
    vs[r][c] = (r < NTOK) ? vb[r * 32 + c] : (_Float16)0.0f;
  }
  __syncthreads();

  int wave = tid >> 5, lane = tid & 31;
  int g = lane >> 4, hl = lane & 15;
  const float* mrow = mask + (size_t)(win & (NWB - 1)) * NTOK * NTOK;

  // --- stage 1: 49 score tiles, one 16x16x32 WMMA each (head_dim == 32) ---
  for (int tile = wave; tile < 49; tile += 8) {
    int tn = tile / 7, tm = tile % 7;
    v16h a, bm;
    { const _Float16* rp = &qs[tn * 16 + hl][0];
#pragma unroll
      for (int e = 0; e < 8; ++e) { a[e] = rp[g * 8 + e]; a[e + 8] = rp[16 + g * 8 + e]; } }
    { const _Float16* rp = &ksm[tm * 16 + hl][0];
#pragma unroll
      for (int e = 0; e < 8; ++e) { bm[e] = rp[g * 8 + e]; bm[e + 8] = rp[16 + g * 8 + e]; } }
    v8f acc = {};
    acc = wmma16(a, bm, acc);
    int m = tm * 16 + hl;
    int zm = m / 49, rm = m % 49, ym = rm / 7, xm = rm % 7;
#pragma unroll
    for (int vv = 0; vv < 8; ++vv) {
      int n = tn * 16 + vv + g * 8;
      float val = acc[vv];
      if (n < NTOK && m < NTOK) {
        int zn = n / 49, rn = n % 49, yn = rn / 7, xn = rn % 7;
        int rpi = ((zn - zm + 1) * 13 + (yn - ym + 6)) * 13 + (xn - xm + 6);
        val += rpb_table[rpi * HEADS + head] + mrow[n * NTOK + m];
      }
      S[n][m] = (_Float16)val;
    }
  }
  __syncthreads();

  // --- stage 2: row softmax (wave per row). Cols >=98 written as exact 0. ---
  for (int r = wave; r < NTOK; r += 8) {
    float vals[4]; float mx = -1e30f;
#pragma unroll
    for (int j = 0; j < 4; ++j) {
      int c = lane + 32 * j;
      float vv = (c < NTOK) ? (float)S[r][c] : -1e30f;
      vals[j] = vv; mx = fmaxf(mx, vv);
    }
#pragma unroll
    for (int off = 16; off; off >>= 1) mx = fmaxf(mx, __shfl_xor(mx, off, 32));
    float sum = 0.0f;
#pragma unroll
    for (int j = 0; j < 4; ++j) {
      int c = lane + 32 * j;
      float e = (c < NTOK) ? __expf(vals[j] - mx) : 0.0f;
      vals[j] = e; sum += e;
    }
#pragma unroll
    for (int off = 16; off; off >>= 1) sum += __shfl_xor(sum, off, 32);
    float rs = 1.0f / sum;
#pragma unroll
    for (int j = 0; j < 4; ++j) S[r][lane + 32 * j] = (_Float16)(vals[j] * rs);
  }
  __syncthreads();

  // --- stage 3: O = P @ v. 7 row tiles x 2 dim tiles, K = 128 (4 steps). ---
  for (int tile = wave; tile < 14; tile += 8) {
    int tn = tile >> 1, td = tile & 1;
    v8f acc = {};
#pragma unroll
    for (int kbs = 0; kbs < 4; ++kbs) {
      v16h a, bm;
      { const _Float16* rp = &S[tn * 16 + hl][kbs * 32];
#pragma unroll
        for (int e = 0; e < 8; ++e) { a[e] = rp[g * 8 + e]; a[e + 8] = rp[16 + g * 8 + e]; } }
#pragma unroll
      for (int e = 0; e < 8; ++e) { bm[e] = vs[kbs * 32 + g * 8 + e][td * 16 + hl]; bm[e + 8] = vs[kbs * 32 + 16 + g * 8 + e][td * 16 + hl]; }
      acc = wmma16(a, bm, acc);
    }
#pragma unroll
    for (int vv = 0; vv < 8; ++vv) {
      int n = tn * 16 + vv + g * 8;
      if (n < NTOK)
        out[((size_t)win * NTOK + n) * CC + head * HDIM + td * 16 + hl] =
            (_Float16)acc[vv];
    }
  }
}

// 5) proj GEMM + window-reverse + inverse roll + residual -> x2 (f32)
__global__ __launch_bounds__(256) void k_proj(const _Float16* __restrict__ ao,
                                              const _Float16* __restrict__ w16,
                                              const float* __restrict__ proj_b,
                                              const float* __restrict__ x,
                                              float* __restrict__ x2) {
  int wave = threadIdx.x >> 5, lane = threadIdx.x & 31;
  int mt = blockIdx.x * 8 + wave;
  int row0 = mt * 16;
  v8f acc[6] = {};
#pragma unroll
  for (int ks = 0; ks < 3; ++ks) {
    v16h a = load_A_g(ao + (size_t)row0 * CC, CC, ks * 32, lane);
#pragma unroll
    for (int nt = 0; nt < 6; ++nt)
      acc[nt] = wmma16(a, load_B_w(w16, 96, nt * 16, ks * 32, lane), acc[nt]);
  }
  int g = lane >> 4, nc = lane & 15;
#pragma unroll
  for (int vv = 0; vv < 8; ++vv) {
    int t = row0 + vv + g * 8;
    int base = win_token_to_addr(t, 0);
#pragma unroll
    for (int nt = 0; nt < 6; ++nt) {
      int c = nt * 16 + nc;
      x2[base + c] = x[base + c] + acc[nt][vv] + proj_b[c];
    }
  }
}

// 6) LN2 over x2 -> f16 rows (natural token order)
__global__ __launch_bounds__(256) void k_ln2(const float* __restrict__ x2,
                                             const float* __restrict__ g2,
                                             const float* __restrict__ b2,
                                             _Float16* __restrict__ h16) {
  int wave = threadIdx.x >> 5, lane = threadIdx.x & 31;
  int t = blockIdx.x * 8 + wave;
  const float* src = x2 + (size_t)t * CC;
  float v0 = src[lane], v1 = src[lane + 32], v2 = src[lane + 64];
  float s = v0 + v1 + v2, ss = v0 * v0 + v1 * v1 + v2 * v2;
#pragma unroll
  for (int off = 16; off; off >>= 1) {
    s  += __shfl_xor(s,  off, 32);
    ss += __shfl_xor(ss, off, 32);
  }
  float mu = s * (1.0f / 96.0f);
  float rinv = rsqrtf(ss * (1.0f / 96.0f) - mu * mu + 1e-5f);
  _Float16* dst = h16 + (size_t)t * CC;
  dst[lane]      = (_Float16)((v0 - mu) * rinv * g2[lane]      + b2[lane]);
  dst[lane + 32] = (_Float16)((v1 - mu) * rinv * g2[lane + 32] + b2[lane + 32]);
  dst[lane + 64] = (_Float16)((v2 - mu) * rinv * g2[lane + 64] + b2[lane + 64]);
}

// 7) fused MLP: FC1(96->384) + GELU + FC2(384->96) + residual, per 16-row tile.
// Hidden split into 4 chunks of 96; GELU output relayout D->A via wave-private
// LDS (same-wave DS ops are in-order, no barrier needed).
__global__ __launch_bounds__(256) void k_mlp(const _Float16* __restrict__ h16,
                                             const _Float16* __restrict__ w1,
                                             const float* __restrict__ fc1_b,
                                             const _Float16* __restrict__ w2,
                                             const float* __restrict__ fc2_b,
                                             const float* __restrict__ x2,
                                             float* __restrict__ out) {
  __shared__ _Float16 stage[8][16][96];
  int wave = threadIdx.x >> 5, lane = threadIdx.x & 31;
  int g = lane >> 4, hl = lane & 15;
  int mt = blockIdx.x * 8 + wave;
  int row0 = mt * 16;

  v16h aX[3];
#pragma unroll
  for (int ks = 0; ks < 3; ++ks)
    aX[ks] = load_A_g(h16 + (size_t)row0 * CC, CC, ks * 32, lane);

  v8f acc2[6] = {};
#pragma unroll
  for (int hc = 0; hc < 4; ++hc) {
    int hbase = hc * 96;
    v8f acc1[6] = {};
#pragma unroll
    for (int ks = 0; ks < 3; ++ks)
#pragma unroll
      for (int nt = 0; nt < 6; ++nt)
        acc1[nt] = wmma16(aX[ks], load_B_w(w1, 96, hbase + nt * 16, ks * 32, lane),
                          acc1[nt]);
    // bias + exact GELU, stage into LDS in row-major per wave slice
#pragma unroll
    for (int nt = 0; nt < 6; ++nt) {
      int c = nt * 16 + hl;
      float bias = fc1_b[hbase + c];
#pragma unroll
      for (int vv = 0; vv < 8; ++vv) {
        float xv = acc1[nt][vv] + bias;
        float gv = 0.5f * xv * (1.0f + erff(xv * 0.70710678118654752f));
        stage[wave][vv + g * 8][c] = (_Float16)gv;
      }
    }
    // FC2 partial accumulation over this 96-wide hidden chunk
#pragma unroll
    for (int ks = 0; ks < 3; ++ks) {
      v16h a;
      { const _Float16* rp = &stage[wave][hl][ks * 32];
#pragma unroll
        for (int e = 0; e < 8; ++e) { a[e] = rp[g * 8 + e]; a[e + 8] = rp[16 + g * 8 + e]; } }
#pragma unroll
      for (int nt = 0; nt < 6; ++nt)
        acc2[nt] = wmma16(a, load_B_w(w2, 384, nt * 16, hbase + ks * 32, lane),
                          acc2[nt]);
    }
  }
  int nc = lane & 15;
#pragma unroll
  for (int vv = 0; vv < 8; ++vv) {
    size_t base = (size_t)(row0 + vv + g * 8) * CC;
#pragma unroll
    for (int nt = 0; nt < 6; ++nt) {
      int c = nt * 16 + nc;
      out[base + c] = x2[base + c] + acc2[nt][vv] + fc2_b[c];
    }
  }
}

// ---------------------------------------------------------------------------
extern "C" void kernel_launch(void* const* d_in, const int* in_sizes, int n_in,
                              void* d_out, int out_size, void* d_ws, size_t ws_size,
                              hipStream_t stream) {
  (void)in_sizes; (void)n_in; (void)out_size; (void)ws_size;
  const float* x         = (const float*)d_in[0];
  const float* attn_mask = (const float*)d_in[1];
  const float* g1        = (const float*)d_in[2];
  const float* b1        = (const float*)d_in[3];
  const float* qkv_w     = (const float*)d_in[4];
  const float* qkv_b     = (const float*)d_in[5];
  const float* rpb_table = (const float*)d_in[6];
  const float* proj_w    = (const float*)d_in[7];
  const float* proj_b    = (const float*)d_in[8];
  const float* g2        = (const float*)d_in[9];
  const float* b2        = (const float*)d_in[10];
  const float* fc1_w     = (const float*)d_in[11];
  const float* fc1_b     = (const float*)d_in[12];
  const float* fc2_w     = (const float*)d_in[13];
  const float* fc2_b     = (const float*)d_in[14];
  float* out = (float*)d_out;

  char* ws = (char*)d_ws;
  _Float16* wq16 = (_Float16*)(ws + OFF_WQKV);
  _Float16* wp16 = (_Float16*)(ws + OFF_WPROJ);
  _Float16* w116 = (_Float16*)(ws + OFF_WFC1);
  _Float16* w216 = (_Float16*)(ws + OFF_WFC2);
  _Float16* xw16 = (_Float16*)(ws + OFF_XW);   // LN1 windows; reused as attn_out
  _Float16* q16  = (_Float16*)(ws + OFF_Q);    // reused as LN2 output
  _Float16* k16  = (_Float16*)(ws + OFF_K);
  _Float16* v16  = (_Float16*)(ws + OFF_V);
  float*    x2   = (float*)(ws + OFF_X2);

  k_cvt<<<144, 256, 0, stream>>>(qkv_w, proj_w, fc1_w, fc2_w, wq16, wp16, w116, w216);
  k_ln1<<<MTOT / 8, 256, 0, stream>>>(x, g1, b1, xw16);
  k_qkv<<<MTILES / 8, 256, 0, stream>>>(xw16, wq16, qkv_b, q16, k16, v16);
  k_attn<<<NWIN * HEADS, 256, 0, stream>>>(q16, k16, v16, rpb_table, attn_mask, xw16);
  k_proj<<<MTILES / 8, 256, 0, stream>>>(xw16, wp16, proj_b, x, x2);
  k_ln2<<<MTOT / 8, 256, 0, stream>>>(x2, g2, b2, q16);
  k_mlp<<<MTILES / 8, 256, 0, stream>>>(q16, w116, fc1_b, w216, fc2_b, x2, out);
}
